// Transition_11811160064193
// MI455X (gfx1250) — matrix operand order repaired and checked by reading, not compiled, not run on hardware
//
#include <hip/hip_runtime.h>
#include <hip/hip_bf16.h>
#include <stdint.h>

// ---------------------------------------------------------------------------
// LSTM (seq=8192, hidden=2048, in=16, out=3+3) for MI455X / gfx1250.
//
//  * Persistent whole-chip kernel, 256 WGs; each WG owns 8 h-indices and its
//    32 W_hh rows resident in LDS (f16, padded stride -> conflict-free b128).
//  * Per step: h broadcast via global_load_async_to_lds_b128 (ASYNCcnt),
//    matvec via v_wmma_f32_16x16x32_f16 with h replicated across B columns,
//    double-buffered A/B registers to overlap ds loads with WMMA.
//  * One grid-wide software barrier per step (monotonic counter+generation,
//    re-zeroed by prep kernel -> deterministic under graph replay).
//  * Final outputs: WMMA GEMM hs[8192,2048] @ Wcat^T[2048,16] + bias.
// ---------------------------------------------------------------------------

typedef __attribute__((ext_vector_type(16))) _Float16 v16h;
typedef __attribute__((ext_vector_type(8)))  _Float16 v8h;
typedef __attribute__((ext_vector_type(8)))  float    v8f;

#define HID    2048
#define SEQ    8192
#define IN_DIM 16
#define NWG    256
#define JPW    8            // h elements per WG
#define ROWS   32           // 4 gates * JPW
#define NTILE  2            // 16-row tiles per WG
#define WSTR   2056         // padded LDS row stride (f16) -> conflict-free b128
#define KC     32           // WMMA K
#define NKC    (HID / KC)   // 64 K-chunks total
#define NCH    (NKC / 4)    // 16 chunks per wave (4-way K split)

union V16U { v16h v; v8h h[2]; };

// workspace layout (bytes)
#define WS_CNT   0                         // unsigned arrival counter
#define WS_GEN   4                         // unsigned generation flag
#define WS_HBUF  256                       // 2 x 2048 f16 double buffer (8 KB)
#define WS_WCT   8704                      // 16 x 2048 f16 packed output weights
#define WS_HS    74240                     // 8192 x 2048 f16 h history (32 MB)

#define WMMA_F16(acc, A, B)                                                   \
  __builtin_amdgcn_wmma_f32_16x16x32_f16(false, (A), false, (B), (short)0,    \
                                         (acc), false, false)

// ---------------------------------------------------------------------------
__global__ void prep_kernel(const float* __restrict__ Wuvw,
                            const float* __restrict__ Wpqr,
                            char* __restrict__ ws) {
  int idx = blockIdx.x * blockDim.x + threadIdx.x;
  _Float16* wct = (_Float16*)(ws + WS_WCT);
  if (idx < 16 * HID) {                    // Wcat^T[n][k], n-major, K contiguous
    int n = idx >> 11, k = idx & (HID - 1);
    float v = 0.f;
    if (n < 3)      v = Wuvw[n * HID + k];
    else if (n < 6) v = Wpqr[(n - 3) * HID + k];
    wct[n * HID + k] = (_Float16)v;
  }
  _Float16* hb = (_Float16*)(ws + WS_HBUF);
  if (idx < 2 * HID) hb[idx] = (_Float16)0.f;   // h_{-1} = 0 (both buffers)
  if (idx == 0) {
    *(unsigned*)(ws + WS_CNT) = 0u;
    *(unsigned*)(ws + WS_GEN) = 0u;
  }
}

// ---------------------------------------------------------------------------
__launch_bounds__(256, 2)
__global__ void lstm_kernel(const float* __restrict__ sa,
                            const float* __restrict__ Wih,
                            const float* __restrict__ Whh,
                            const float* __restrict__ bih,
                            const float* __restrict__ bhh,
                            char* __restrict__ ws) {
  extern __shared__ char smem[];
  _Float16* wlds  = (_Float16*)smem;                           // 32*2056 f16
  _Float16* hlds  = (_Float16*)(smem + ROWS * WSTR * 2);       // 2048 f16
  float*    red   = (float*)(smem + ROWS * WSTR * 2 + HID * 2);// [2][4][16]
  float*    gatesL= red + NTILE * 4 * 16;                      // [32]

  const int tid  = threadIdx.x;
  const int wg   = blockIdx.x;
  const int j0   = wg * JPW;
  const int lane = tid & 31;
  const int wid  = tid >> 5;
  const int tile = wid >> 2;       // 0..1 : which 16-row tile
  const int ksl  = wid & 3;        // 0..3 : K slice
  const int m    = lane & 15;      // row within tile (A) / column (B,D)
  const int half = lane >> 4;      // K-half selector per ISA layouts

  unsigned* cnt  = (unsigned*)(ws + WS_CNT);
  unsigned* gen  = (unsigned*)(ws + WS_GEN);
  _Float16* hbuf = (_Float16*)(ws + WS_HBUF);
  _Float16* hs   = (_Float16*)(ws + WS_HS);

  // LDS byte offset of this thread's 16B slot in hlds (flat LDS addr low bits)
  const unsigned hldsOff = (unsigned)(uintptr_t)(hlds) + (unsigned)tid * 16u;

  // ---- stage resident W_hh slice into LDS (f32 -> f16), tile order i,f | g,o
  for (int idx = tid; idx < ROWS * HID; idx += 256) {
    int r = idx >> 11, k = idx & (HID - 1);
    int grow = (r >> 3) * HID + j0 + (r & 7);          // gate*2048 + j
    wlds[r * WSTR + k] = (_Float16)Whh[(size_t)grow * HID + k];
  }
  // ---- per-thread input-projection row (threads 0..31 own one gate row)
  float wih[IN_DIM]; float bsum = 0.f;
  if (tid < ROWS) {
    int grow = (tid >> 3) * HID + j0 + (tid & 7);
#pragma unroll
    for (int q = 0; q < IN_DIM; ++q) wih[q] = Wih[grow * IN_DIM + q];
    bsum = bih[grow] + bhh[grow];
  }
  float c = 0.f;                                        // cell state (tid<8)
  __syncthreads();

  for (int t = 0; t < SEQ; ++t) {
    // ---- async broadcast h_{t-1} global -> LDS (16B per lane, ASYNCcnt)
    {
      const _Float16* gsrc = hbuf + (t & 1) * HID + tid * 8;
      asm volatile("global_load_async_to_lds_b128 %0, %1, off"
                   :: "v"(hldsOff), "v"((uint64_t)(uintptr_t)gsrc)
                   : "memory");
      asm volatile("s_wait_asynccnt 0x0" ::: "memory");
    }
    __syncthreads();

    // ---- matvec: 16 WMMAs per wave, double-buffered A/B to hide LDS latency
    v8f acc = {};
    const _Float16* wbase = wlds + (tile * 16 + m) * WSTR;
    const _Float16* hbase = hlds + half * 8;
    const int base = ksl * NCH;
    V16U a0, b0, a1, b1;
    {
      int k = base * KC;
      a0.h[0] = *(const v8h*)(wbase + k + half * 8);
      a0.h[1] = *(const v8h*)(wbase + k + 16 + half * 8);
      b0.h[0] = *(const v8h*)(hbase + k);
      b0.h[1] = *(const v8h*)(hbase + k + 16);
    }
#pragma unroll
    for (int cc = 0; cc < NCH; cc += 2) {
      {
        int k = (base + cc + 1) * KC;                   // prefetch chunk cc+1
        a1.h[0] = *(const v8h*)(wbase + k + half * 8);
        a1.h[1] = *(const v8h*)(wbase + k + 16 + half * 8);
        b1.h[0] = *(const v8h*)(hbase + k);
        b1.h[1] = *(const v8h*)(hbase + k + 16);
      }
      acc = WMMA_F16(acc, a0.v, b0.v);                  // consume chunk cc
      if (cc + 2 < NCH) {
        int k = (base + cc + 2) * KC;                   // prefetch chunk cc+2
        a0.h[0] = *(const v8h*)(wbase + k + half * 8);
        a0.h[1] = *(const v8h*)(wbase + k + 16 + half * 8);
        b0.h[0] = *(const v8h*)(hbase + k);
        b0.h[1] = *(const v8h*)(hbase + k + 16);
      }
      acc = WMMA_F16(acc, a1.v, b1.v);                  // consume chunk cc+1
    }
    // every D column identical; lane0 -> rows 0..7, lane16 -> rows 8..15
    if (m == 0) {
#pragma unroll
      for (int r = 0; r < 8; ++r)
        red[(tile * 4 + ksl) * 16 + half * 8 + r] = acc[r];
    }
    __syncthreads();

    // ---- combine K-slices + on-the-fly input projection
    if (tid < ROWS) {
      int tl = tid >> 4, mm = tid & 15;
      float g4 = red[(tl * 4 + 0) * 16 + mm] + red[(tl * 4 + 1) * 16 + mm]
               + red[(tl * 4 + 2) * 16 + mm] + red[(tl * 4 + 3) * 16 + mm];
      const float* x = sa + (size_t)t * IN_DIM;
      float xp = bsum;
#pragma unroll
      for (int q = 0; q < IN_DIM; ++q) xp += wih[q] * x[q];
      gatesL[tid] = g4 + xp;
    }
    __syncthreads();

    // ---- activations + state update, publish h_new
    if (tid < JPW) {
      float ig = 1.f / (1.f + __expf(-gatesL[tid]));
      float fg = 1.f / (1.f + __expf(-gatesL[8 + tid]));
      float gg = tanhf(gatesL[16 + tid]);
      float og = 1.f / (1.f + __expf(-gatesL[24 + tid]));
      c = fg * c + ig * gg;
      float hn = og * tanhf(c);
      _Float16 h16 = (_Float16)hn;
      hbuf[((t + 1) & 1) * HID + j0 + tid] = h16;
      hs[(size_t)t * HID + j0 + tid] = h16;
    }
    __syncthreads();

    // ---- grid-wide barrier (monotonic count, generation flag)
    if (tid == 0) {
      __threadfence();
      unsigned arrived = __hip_atomic_fetch_add(cnt, 1u, __ATOMIC_ACQ_REL,
                                                __HIP_MEMORY_SCOPE_AGENT) + 1u;
      unsigned target = (unsigned)NWG * (unsigned)(t + 1);
      if (arrived == target) {
        __hip_atomic_store(gen, (unsigned)(t + 1), __ATOMIC_RELEASE,
                           __HIP_MEMORY_SCOPE_AGENT);
      } else {
        while (__hip_atomic_load(gen, __ATOMIC_ACQUIRE,
                                 __HIP_MEMORY_SCOPE_AGENT) < (unsigned)(t + 1))
          __builtin_amdgcn_s_sleep(1);
      }
    }
    __syncthreads();
  }
}

// ---------------------------------------------------------------------------
// out[8192,6] = hs[8192,2048](f16) @ Wcat^T[2048,16](f16, cols 0..5) + bias
__global__ void out_gemm(const char* __restrict__ ws,
                         const float* __restrict__ buvw,
                         const float* __restrict__ bpqr,
                         float* __restrict__ out) {
  const _Float16* hs  = (const _Float16*)(ws + WS_HS);
  const _Float16* wct = (const _Float16*)(ws + WS_WCT);
  const int t0   = blockIdx.x * 16;
  const int lane = threadIdx.x & 31;
  const int m    = lane & 15;
  const int half = lane >> 4;

  v8f acc = {};
  const _Float16* abase = hs + (size_t)(t0 + m) * HID;   // A row = timestep
  const _Float16* bbase = wct + m * HID + half * 8;      // B col n = m
  V16U a0, b0, a1, b1;
  a0.h[0] = *(const v8h*)(abase + half * 8);
  a0.h[1] = *(const v8h*)(abase + 16 + half * 8);
  b0.h[0] = *(const v8h*)(bbase);
  b0.h[1] = *(const v8h*)(bbase + 16);
#pragma unroll 4
  for (int cc = 0; cc < NKC; cc += 2) {
    {
      int k = (cc + 1) * KC;                             // prefetch chunk cc+1
      a1.h[0] = *(const v8h*)(abase + k + half * 8);
      a1.h[1] = *(const v8h*)(abase + k + 16 + half * 8);
      b1.h[0] = *(const v8h*)(bbase + k);
      b1.h[1] = *(const v8h*)(bbase + k + 16);
    }
    acc = WMMA_F16(acc, a0.v, b0.v);
    if (cc + 2 < NKC) {
      int k = (cc + 2) * KC;                             // prefetch chunk cc+2
      a0.h[0] = *(const v8h*)(abase + k + half * 8);
      a0.h[1] = *(const v8h*)(abase + k + 16 + half * 8);
      b0.h[0] = *(const v8h*)(bbase + k);
      b0.h[1] = *(const v8h*)(bbase + k + 16);
    }
    acc = WMMA_F16(acc, a1.v, b1.v);
  }
  if (m < 6) {
    float bias = (m < 3) ? buvw[m] : bpqr[m - 3];
#pragma unroll
    for (int r = 0; r < 8; ++r) {
      int M = r + (half ? 8 : 0);                        // D row layout
      out[(size_t)(t0 + M) * 6 + m] = acc[r] + bias;
    }
  }
}

// ---------------------------------------------------------------------------
extern "C" void kernel_launch(void* const* d_in, const int* in_sizes, int n_in,
                              void* d_out, int out_size, void* d_ws, size_t ws_size,
                              hipStream_t stream) {
  const float* sa   = (const float*)d_in[0];  // [8192,16]
  const float* Wih  = (const float*)d_in[1];  // [8192,16]
  const float* Whh  = (const float*)d_in[2];  // [8192,2048]
  const float* bih  = (const float*)d_in[3];  // [8192]
  const float* bhh  = (const float*)d_in[4];  // [8192]
  const float* Wuvw = (const float*)d_in[5];  // [3,2048]
  const float* buvw = (const float*)d_in[6];  // [3]
  const float* Wpqr = (const float*)d_in[7];  // [3,2048]
  const float* bpqr = (const float*)d_in[8];  // [3]
  char*  ws  = (char*)d_ws;
  float* out = (float*)d_out;

  prep_kernel<<<128, 256, 0, stream>>>(Wuvw, Wpqr, ws);

  size_t lds = (size_t)ROWS * WSTR * 2 + HID * 2 + NTILE * 4 * 16 * 4 + ROWS * 4;
  lstm_kernel<<<NWG, 256, lds, stream>>>(sa, Wih, Whh, bih, bhh, ws);

  out_gemm<<<SEQ / 16, 32, 0, stream>>>(ws, buvw, bpqr, out);
}
